// NSLM_84902913507935
// MI455X (gfx1250) — matrix operand order repaired and checked by reading, not compiled, not run on hardware
//
#include <hip/hip_runtime.h>

// ---------------------------------------------------------------------------
// MI455X (gfx1250, wave32).
// Precision: bf16 A/B, f32 WMMA accumulation (v_wmma_f32_16x16x32_bf16).
// 42 GFLOP total / ~120MB moved: bf16 WMMA puts compute at/under the ~5us
// HBM time; f32 WMMA (16x16x4) would be ~8x more matrix instructions.
// New this round: B-panels staged in LDS via the Tensor Data Mover
// (tensor_load_to_lds + s_wait_tensorcnt), TDM pad fields chosen so column
// reads are LDS bank-conflict-free; 32x64 per-wave tiles (8 accumulators).
// ---------------------------------------------------------------------------

typedef __bf16 bf16;
typedef __attribute__((ext_vector_type(16))) __bf16 v16bf;
typedef __attribute__((ext_vector_type(8)))  __bf16 v8bf;
typedef __attribute__((ext_vector_type(4)))  __bf16 v4bf;
typedef __attribute__((ext_vector_type(8)))  float  v8f;
typedef __attribute__((ext_vector_type(2)))  float  v2f;
typedef __attribute__((ext_vector_type(4)))  unsigned int u32x4;
typedef __attribute__((ext_vector_type(8)))  int i32x8;
typedef __attribute__((ext_vector_type(4)))  int i32x4;

#define BB 4
#define LL 512
#define HH 768
#define EE 10
#define RR 10
#define MM (BB * LL)   // 2048

// LDS panel: 64 B-rows x 768 k (bf16). TDM pads 4 dwords every 128 payload
// dwords -> row stride 396 dwords (1584B). 396 % 64 = 12 -> the 16 lanes of a
// b128 column read hit disjoint bank quads (12n mod 64 distinct).
#define LDS_ROW_BYTES 1584
#define PANEL_LDS_BYTES (64 * LDS_ROW_BYTES)   // 101376 B (<320KB/WGP; 3 blocks/WGP)

static __device__ inline unsigned lds_off(int row, int k) {
  // byte offset of element (row, k) in the padded LDS panel
  return (unsigned)(row * LDS_ROW_BYTES + k * 2 + ((k >> 8) << 4));
}

static __device__ inline v8f zero8() {
  v8f z = {0.f, 0.f, 0.f, 0.f, 0.f, 0.f, 0.f, 0.f};
  return z;
}

// A-fragment (16x32 bf16, row-major, lda elems/row). Lanes 0-15 hold row
// M=lane, K=[k0,k0+16); lanes 16-31 hold K=[k0+16,k0+32) (matches B).
// -> two global b128 loads.
static __device__ inline v16bf load_a_frag(const bf16* __restrict__ A, int lda,
                                           int m0, int k0, int lane) {
  const int row = m0 + (lane & 15);
  const int kb  = k0 + ((lane >> 4) << 4);
  const bf16* p = A + (size_t)row * lda + kb;
  v8bf lo = *reinterpret_cast<const v8bf*>(p);
  v8bf hi = *reinterpret_cast<const v8bf*>(p + 8);
  v16bf f;
#pragma unroll
  for (int i = 0; i < 8; ++i) { f[i] = lo[i]; f[8 + i] = hi[i]; }
  return f;
}

// B-fragment (32x16) from global row-major (N,K) weight: lanes 0-15 col n
// hold K=[k0,k0+16), lanes 16-31 K=[k0+16,k0+32) -> one contiguous 32B load.
static __device__ inline v16bf load_b_frag(const bf16* __restrict__ W, int ldb,
                                           int n0, int k0, int lane) {
  const int col = n0 + (lane & 15);
  const int kb  = k0 + ((lane >> 4) << 4);
  return *reinterpret_cast<const v16bf*>(W + (size_t)col * ldb + kb);
}

// Same fragment but from the padded LDS panel (2x ds_load_b128, 16B aligned).
static __device__ inline v16bf lds_b_frag(const char* smem, int n_local,
                                          int k0, int lane) {
  const int col = n_local + (lane & 15);
  const int kb  = k0 + ((lane >> 4) << 4);
  return *reinterpret_cast<const v16bf*>(smem + lds_off(col, kb));
}

// ---------------------------------------------------------------------------
// Stage a 64-row x 768-k bf16 panel of W (rows n0..n0+63) into LDS.
// Preferred path: one TDM descriptor issued by wave 0 (D# per ISA 8.3/8.4),
// tracked with TENSORcnt.  Fallback: cooperative load + ds_store.
// ---------------------------------------------------------------------------
static __device__ inline void load_panel(char* smem, const bf16* __restrict__ W,
                                         int n0) {
#if __has_builtin(__builtin_amdgcn_tensor_load_to_lds)
  if (threadIdx.x < 32) {   // one wave issues the DMA; EXEC is ignored by TDM
    unsigned long long ga =
        (unsigned long long)(const void*)(W + (size_t)n0 * HH);
    // Flat LDS aperture: addr[31:0] is the LDS byte offset (ISA 10.2).
    unsigned lbase = (unsigned)(unsigned long long)(const void*)smem;
    // Group0: count=1 | lds_addr | global_addr[56:0] | type=2 (bits 127:126)
    u32x4 g0 = { 1u, lbase,
                 (unsigned)(ga & 0xffffffffull),
                 (unsigned)((ga >> 32) & 0x01ffffffull) | 0x80000000u };
    // Group1: data_size=2B(b17:16=1), pad_enable(b20), pad_interval=128dw
    // (code 6, b24:22), pad_amount=4dw (code 3, b31:25)  -> 0x07910000
    //   d1[31:16]=tensor_dim0=768, d2[31:16]=tensor_dim1=768,
    //   d3[31:16]=tile_dim0=768, d4[15:0]=tile_dim1=64, d5=dim0_stride=768
    i32x8 g1 = { (int)0x07910000u,
                 (int)0x03000000u,
                 (int)0x03000000u,
                 (int)0x03000000u,
                 64,
                 (int)HH,
                 0, 0 };
    i32x4 g2 = {};   // 2D tensor: groups 2/3 unused
    i32x4 g3 = {};
#if defined(__clang_major__) && __clang_major__ >= 23
    i32x8 g4 = {};
    __builtin_amdgcn_tensor_load_to_lds(g0, g1, g2, g3, g4, 0);
#else
    __builtin_amdgcn_tensor_load_to_lds(g0, g1, g2, g3, 0);
#endif
#if __has_builtin(__builtin_amdgcn_s_wait_tensorcnt)
    __builtin_amdgcn_s_wait_tensorcnt(0);
#endif
  }
#else
  // Cooperative fallback: 128 threads, 16B chunks, same padded layout.
  for (int c = threadIdx.x; c < 64 * (HH / 8); c += 128) {
    const int row = c / (HH / 8);
    const int k   = (c - row * (HH / 8)) * 8;
    v8bf v = *reinterpret_cast<const v8bf*>(W + (size_t)(n0 + row) * HH + k);
    *reinterpret_cast<v8bf*>(smem + lds_off(row, k)) = v;
  }
#endif
  __syncthreads();
}

// ---------------------------------------------------------------------------
// fp32 -> bf16 conversion, 4 elements/thread
// ---------------------------------------------------------------------------
__global__ void __launch_bounds__(256)
cvt_f32_bf16(const float* __restrict__ src, bf16* __restrict__ dst, int n4) {
  int i = blockIdx.x * blockDim.x + threadIdx.x;
  if (i >= n4) return;
  float4 v = reinterpret_cast<const float4*>(src)[i];
  v4bf o = { (bf16)v.x, (bf16)v.y, (bf16)v.z, (bf16)v.w };
  reinterpret_cast<v4bf*>(dst)[i] = o;
}

// ---------------------------------------------------------------------------
// Panel GEMM: out[r] = A @ W[r]^T, optional bias+relu epilogue (bf16 out).
// Block: 4 waves, each 32(M)x64(N) (8 accumulators); block tile 128x64.
// B-panel (64 rows x 768 k) staged in LDS once per block via TDM.
// grid = (HH/64, MM/128, nR)
// ---------------------------------------------------------------------------
template <bool BIAS_RELU>
__global__ void __launch_bounds__(128)
gemm_panel(const bf16* __restrict__ A, const bf16* __restrict__ W,
           const float* __restrict__ bias, bf16* __restrict__ out) {
  extern __shared__ char smem[];
  const int lane = threadIdx.x & 31;
  const int wave = threadIdx.x >> 5;
  const int n0 = blockIdx.x * 64;
  const int m0 = blockIdx.y * 128 + wave * 32;
  const int r  = blockIdx.z;
  const bf16* __restrict__ Wr = W + (size_t)r * HH * HH;
  bf16* __restrict__ Or       = out + (size_t)r * MM * HH;

  load_panel(smem, Wr, n0);

  v8f acc[2][4];
#pragma unroll
  for (int mi = 0; mi < 2; ++mi)
#pragma unroll
    for (int j = 0; j < 4; ++j) acc[mi][j] = zero8();

  for (int k0 = 0; k0 < HH; k0 += 32) {
    v16bf a0 = load_a_frag(A, HH, m0,      k0, lane);
    v16bf a1 = load_a_frag(A, HH, m0 + 16, k0, lane);
#pragma unroll
    for (int j = 0; j < 4; ++j) {
      v16bf b = lds_b_frag(smem, 16 * j, k0, lane);
      acc[0][j] = __builtin_amdgcn_wmma_f32_16x16x32_bf16(
          false, a0, false, b, (short)0, acc[0][j], false, false);
      acc[1][j] = __builtin_amdgcn_wmma_f32_16x16x32_bf16(
          false, a1, false, b, (short)0, acc[1][j], false, false);
    }
  }

  // C/D layout: lane<16 -> col=lane, rows +0..7 in v0..7; lane>=16 -> rows +8.
  const int cn = lane & 15;
#pragma unroll
  for (int mi = 0; mi < 2; ++mi) {
    const int rbase = m0 + mi * 16 + ((lane >> 4) << 3);
#pragma unroll
    for (int j = 0; j < 4; ++j) {
      const int col = n0 + 16 * j + cn;
      const float bv = BIAS_RELU ? bias[col] : 0.f;
#pragma unroll
      for (int i = 0; i < 8; ++i) {
        float v = acc[mi][j][i] + bv;
        if (BIAS_RELU) v = v > 0.f ? v : 0.f;
        Or[(size_t)(rbase + i) * HH + col] = (bf16)v;
      }
    }
  }
}

// ---------------------------------------------------------------------------
// ent_logits = h_proj @ W_ent^T + b_ent  (2048x768 · 768x10) — tiny, VALU.
// ---------------------------------------------------------------------------
__global__ void __launch_bounds__(256)
ent_kernel(const bf16* __restrict__ hp, const float* __restrict__ W_ent,
           const float* __restrict__ b_ent, float* __restrict__ out) {
  int idx = blockIdx.x * blockDim.x + threadIdx.x;
  if (idx >= MM * EE) return;
  const int m = idx / EE;
  const int e = idx - m * EE;
  const bf16*  hr = hp + (size_t)m * HH;
  const float* wr = W_ent + (size_t)e * HH;
  float s = b_ent[e];
#pragma unroll 4
  for (int k = 0; k < HH; ++k) s += (float)hr[k] * wr[k];
  out[idx] = s;
}

// ---------------------------------------------------------------------------
// rel[b,l,m,r] = h_proj[b,l,:] . t[r][b,m,:] + b_bil[r]
// Per-wave: one 16x16 (l,m) tile with 10 accumulators (one per r); A frag
// amortized over 10 WMMAs/k-step; each lane holds all 10 r-values per (l,m)
// element -> contiguous 40B stores (5x float2, 8B aligned). t is L2-resident.
// ---------------------------------------------------------------------------
__global__ void __launch_bounds__(256)
gemm_rel(const bf16* __restrict__ hp, const bf16* __restrict__ t,
         const float* __restrict__ b_bil, float* __restrict__ out_rel) {
  const int lane = threadIdx.x & 31;
  const int wave = threadIdx.x >> 5;
  const int b    = blockIdx.y;
  const int tile = blockIdx.x * 8 + wave;     // 0..1023 : 32x32 (l,m) tiles
  const int l0 = (tile >> 5) << 4;
  const int m0 = (tile & 31) << 4;
  const bf16* __restrict__ A = hp + (size_t)b * LL * HH;

  v8f acc[RR];
#pragma unroll
  for (int r = 0; r < RR; ++r) acc[r] = zero8();

  for (int k0 = 0; k0 < HH; k0 += 32) {
    v16bf a = load_a_frag(A, HH, l0, k0, lane);
#pragma unroll
    for (int r = 0; r < RR; ++r) {
      const bf16* __restrict__ Tr = t + ((size_t)r * MM + (size_t)b * LL) * HH;
      v16bf bf = load_b_frag(Tr, HH, m0, k0, lane);
      acc[r] = __builtin_amdgcn_wmma_f32_16x16x32_bf16(
          false, a, false, bf, (short)0, acc[r], false, false);
    }
  }

  float bias[RR];
#pragma unroll
  for (int r = 0; r < RR; ++r) bias[r] = b_bil[r];

  const int cm    = m0 + (lane & 15);
  const int rbase = l0 + ((lane >> 4) << 3);
#pragma unroll
  for (int i = 0; i < 8; ++i) {
    const int l = rbase + i;
    float* p = out_rel + (((size_t)b * LL + l) * LL + cm) * RR;  // 8B aligned
#pragma unroll
    for (int r = 0; r < RR; r += 2) {
      v2f v = { acc[r][i] + bias[r], acc[r + 1][i] + bias[r + 1] };
      *reinterpret_cast<v2f*>(p + r) = v;
    }
  }
}

// ---------------------------------------------------------------------------
// Host-side launch.  Workspace (~50.7 MB): hb, wpb, wbb, hp, t (all bf16).
// ---------------------------------------------------------------------------
extern "C" void kernel_launch(void* const* d_in, const int* in_sizes, int n_in,
                              void* d_out, int out_size, void* d_ws, size_t ws_size,
                              hipStream_t stream) {
  (void)in_sizes; (void)n_in; (void)out_size; (void)ws_size;

  const float* h      = (const float*)d_in[0];
  const float* W_proj = (const float*)d_in[1];
  const float* b_proj = (const float*)d_in[2];
  const float* W_ent  = (const float*)d_in[3];
  const float* b_ent  = (const float*)d_in[4];
  const float* W_bil  = (const float*)d_in[5];
  const float* b_bil  = (const float*)d_in[6];
  float* out = (float*)d_out;

  char* ws = (char*)d_ws;
  bf16* hb  = (bf16*)ws;  ws += (size_t)MM * HH * sizeof(bf16);
  bf16* wpb = (bf16*)ws;  ws += (size_t)HH * HH * sizeof(bf16);
  bf16* wbb = (bf16*)ws;  ws += (size_t)RR * HH * HH * sizeof(bf16);
  bf16* hp  = (bf16*)ws;  ws += (size_t)MM * HH * sizeof(bf16);
  bf16* t   = (bf16*)ws;

  // 1) downconvert inputs to bf16
  {
    int n4 = MM * HH / 4;
    cvt_f32_bf16<<<(n4 + 255) / 256, 256, 0, stream>>>(h, hb, n4);
  }
  {
    int n4 = HH * HH / 4;
    cvt_f32_bf16<<<(n4 + 255) / 256, 256, 0, stream>>>(W_proj, wpb, n4);
  }
  {
    int n4 = RR * HH * HH / 4;
    cvt_f32_bf16<<<(n4 + 255) / 256, 256, 0, stream>>>(W_bil, wbb, n4);
  }

  // 2) h_proj = relu(h @ W_proj^T + b_proj)
  gemm_panel<true><<<dim3(HH / 64, MM / 128, 1), 128, PANEL_LDS_BYTES, stream>>>(
      hb, wpb, b_proj, hp);

  // 3) ent_logits (first 20480 floats of d_out)
  ent_kernel<<<(MM * EE + 255) / 256, 256, 0, stream>>>(hp, W_ent, b_ent, out);

  // 4) t[r] = h_proj @ W_bil[r]^T
  gemm_panel<false><<<dim3(HH / 64, MM / 128, RR), 128, PANEL_LDS_BYTES, stream>>>(
      hp, wbb, (const float*)nullptr, t);

  // 5) rel_logits (remaining 10485760 floats of d_out)
  gemm_rel<<<dim3((LL / 16) * (LL / 16) / 8, BB), 256, 0, stream>>>(
      hp, t, b_bil, out + (size_t)BB * LL * EE);
}
